// Model_43130061587000
// MI455X (gfx1250) — matrix operand order repaired and checked by reading, not compiled, not run on hardware
//
#include <hip/hip_runtime.h>
#include <math.h>
#include <stdint.h>

// ---------------------------------------------------------------------------
// CDNA5 (gfx1250) transformer forward pass.
// GEMMs: bf16 operands staged to LDS with global_load_async_to_lds_b128
// (double buffered, ASYNCcnt), compute with v_wmma_f32_16x16x32_bf16.
// ---------------------------------------------------------------------------

typedef __attribute__((ext_vector_type(16))) __bf16 v16bf;
typedef __attribute__((ext_vector_type(8)))  float  v8f;

#define FLAG_TRANSB 1
#define FLAG_RELU   2
#define FLAG_BIAS   4
#define FLAG_PERM   8    // store C[m,n] -> out[b, n, s] with m = b*128+s (logits)
#define FLAG_CBF16  16   // store C as bf16 (feeds another GEMM)

__device__ __forceinline__ unsigned short f2bf(float f) {
  union { float f; unsigned int u; } cv; cv.f = f;
  unsigned int u = cv.u;
  u += 0x7FFFu + ((u >> 16) & 1u);   // round-to-nearest-even
  return (unsigned short)(u >> 16);
}

union Frag { unsigned int d[8]; v16bf v; };

// Async copy of one 16-byte chunk per lane: global -> LDS (ASYNCcnt tracked).
__device__ __forceinline__ void async_ld16(unsigned lds_off, const unsigned short* g) {
  asm volatile("global_load_async_to_lds_b128 %0, %1, off"
               :: "v"(lds_off), "v"((unsigned long long)(size_t)g)
               : "memory");
}
__device__ __forceinline__ void wait_async0() {
  asm volatile("s_wait_asynccnt 0x0" ::: "memory");
}

// Generic batched GEMM:  C = act(alpha * A @ op(B) + bias)
// A: bf16 [M,K] row-major. B: bf16 [K,N] or (FLAG_TRANSB) [N,K].
// Two batch dims: z = b2*nb1 + b1, independent strides (in elements).
// Block tile 64x64, K-tile 32, 128 threads = 4 waves, each wave 32x32 (2x2 WMMA).
__global__ __launch_bounds__(128) void gemm_wmma_bf16(
    const unsigned short* __restrict__ A, const unsigned short* __restrict__ B,
    void* __restrict__ C, const float* __restrict__ bias,
    int M, int N, int K, int lda, int ldb, int ldc,
    long sA1, long sB1, long sC1, int nb1,
    long sA2, long sB2, long sC2,
    float alpha, int flags)
{
  // Row strides padded so every 16B chunk is 16B aligned:
  // As: [64][40] (80B rows).  Bs transB: [64][40].  Bs normal: [32][72] (144B rows).
  __shared__ unsigned short As[2][64 * 40];
  __shared__ unsigned short Bs[2][64 * 40];

  const int z  = blockIdx.z;
  const int b1 = z % nb1, b2 = z / nb1;
  const unsigned short* Ap = A + (long)b1 * sA1 + (long)b2 * sA2;
  const unsigned short* Bp = B + (long)b1 * sB1 + (long)b2 * sB2;

  const int n0 = blockIdx.x * 64;
  const int m0 = blockIdx.y * 64;
  const int tid  = threadIdx.x;
  const int lane = tid & 31;
  const int wave = tid >> 5;
  const int wr = wave >> 1, wc = wave & 1;

  const bool tb = (flags & FLAG_TRANSB) != 0;

  // Issue the 16 async B128 loads (8KB: A 4KB + B 4KB) for one k-tile.
  auto stage = [&](int buf, int k0) {
#pragma unroll
    for (int i = 0; i < 4; ++i) {
      const int part = wave * 4 + i;            // 0..15
      if (part < 8) {                           // A tile: 64 rows x 64B
        const int row = part * 8 + (lane >> 2);
        const int ch  = lane & 3;
        async_ld16((unsigned)(size_t)&As[buf][row * 40 + ch * 8],
                   Ap + (long)(m0 + row) * lda + k0 + ch * 8);
      } else if (tb) {                          // B [N,K]: rows are n, 64B each
        const int row = (part - 8) * 8 + (lane >> 2);
        const int ch  = lane & 3;
        async_ld16((unsigned)(size_t)&Bs[buf][row * 40 + ch * 8],
                   Bp + (long)(n0 + row) * ldb + k0 + ch * 8);
      } else {                                  // B [K,N]: rows are k, 128B each
        const int row = (part - 8) * 4 + (lane >> 3);
        const int ch  = lane & 7;
        async_ld16((unsigned)(size_t)&Bs[buf][row * 72 + ch * 8],
                   Bp + (long)(k0 + row) * ldb + n0 + ch * 8);
      }
    }
  };

  v8f acc[2][2] = {};
  const int nkt = K >> 5;

  stage(0, 0);
  for (int t = 0; t < nkt; ++t) {
    const int buf = t & 1;
    wait_async0();
    __syncthreads();
    if (t + 1 < nkt) stage(buf ^ 1, (t + 1) << 5);   // overlap DMA with WMMA

    // A fragments: 16x32 bf16. lanes0-15: M rows, K 0..15;
    // lanes16-31: same rows, K 16..31; bf16 pairs per dword.
    Frag af[2], bf[2];
#pragma unroll
    for (int mt = 0; mt < 2; ++mt) {
      const int mrow = wr * 32 + mt * 16 + (lane & 15);
      const int kb   = (lane >> 4) << 4;
#pragma unroll
      for (int j = 0; j < 8; ++j) {
        const int k = kb + (j << 1);
        af[mt].d[j] = *(const unsigned int*)&As[buf][mrow * 40 + k];
      }
    }
    // B fragments: 32x16. N = lane%16; lanes0-15 K=0..15, lanes16-31 K=16..31.
#pragma unroll
    for (int nt = 0; nt < 2; ++nt) {
      const int ncol = wc * 32 + nt * 16 + (lane & 15);
      const int kg   = (lane >> 4) << 4;
      if (tb) {
#pragma unroll
        for (int j = 0; j < 8; ++j)
          bf[nt].d[j] = *(const unsigned int*)&Bs[buf][ncol * 40 + kg + (j << 1)];
      } else {
#pragma unroll
        for (int j = 0; j < 8; ++j) {
          const unsigned lo = Bs[buf][(kg + (j << 1)) * 72 + ncol];
          const unsigned hi = Bs[buf][(kg + (j << 1) + 1) * 72 + ncol];
          bf[nt].d[j] = lo | (hi << 16);
        }
      }
    }

#pragma unroll
    for (int mt = 0; mt < 2; ++mt)
#pragma unroll
      for (int nt = 0; nt < 2; ++nt)
        acc[mt][nt] = __builtin_amdgcn_wmma_f32_16x16x32_bf16(
            false, af[mt].v, false, bf[nt].v, (short)0, acc[mt][nt], false, false);
    __syncthreads();   // all waves done reading buf before it is restaged
  }

  // Epilogue. D layout: VGPR r: lanes0-15 -> M=r, lanes16-31 -> M=8+r; N=lane%16.
#pragma unroll
  for (int mt = 0; mt < 2; ++mt) {
#pragma unroll
    for (int nt = 0; nt < 2; ++nt) {
      const int n     = n0 + wc * 32 + nt * 16 + (lane & 15);
      const int mbase = m0 + wr * 32 + mt * 16 + ((lane >> 4) << 3);
#pragma unroll
      for (int r = 0; r < 8; ++r) {
        float v = acc[mt][nt][r] * alpha;
        if (flags & FLAG_BIAS) v += bias[n];
        if (flags & FLAG_RELU) v = fmaxf(v, 0.0f);
        const int m = mbase + r;
        if (flags & FLAG_CBF16) {
          ((unsigned short*)C)[(long)b1 * sC1 + (long)b2 * sC2 + (long)m * ldc + n] = f2bf(v);
        } else if (flags & FLAG_PERM) {
          const int b = m >> 7, s = m & 127;
          ((float*)C)[(long)b * N * 128 + (long)n * 128 + s] = v;
        } else {
          ((float*)C)[(long)b1 * sC1 + (long)b2 * sC2 + (long)m * ldc + n] = v;
        }
      }
    }
  }
}

// fp32 -> bf16 bulk convert. n must be a multiple of 1024; grid.x = n/1024.
__global__ __launch_bounds__(256) void cvt_f32_bf16(
    const float* __restrict__ in, unsigned short* __restrict__ out) {
  const long i = (long)blockIdx.x * 1024 + (long)threadIdx.x * 4;
  union { unsigned short u[4]; uint2 v; } r;
#pragma unroll
  for (int j = 0; j < 4; ++j) r.u[j] = f2bf(in[i + j]);
  *(uint2*)(out + i) = r.v;
}

// Row softmax over length-128 rows; reads f32, writes bf16 (for the PV GEMM).
__global__ __launch_bounds__(128) void softmax128(
    const float* __restrict__ S, unsigned short* __restrict__ P, int causal) {
  __shared__ float red[128];
  const int row = blockIdx.x;          // (h*B+b)*128 + s
  const int r   = row & 127;
  const int t   = threadIdx.x;
  float v = S[(long)row * 128 + t];
  if (causal && t > r) v += -1e9f;
  red[t] = v; __syncthreads();
  for (int off = 64; off > 0; off >>= 1) {
    if (t < off) red[t] = fmaxf(red[t], red[t + off]);
    __syncthreads();
  }
  const float mx = red[0]; __syncthreads();
  const float e = __expf(v - mx);
  red[t] = e; __syncthreads();
  for (int off = 64; off > 0; off >>= 1) {
    if (t < off) red[t] += red[t + off];
    __syncthreads();
  }
  P[(long)row * 128 + t] = f2bf(e / red[0]);
}

// y = LayerNorm(x + a)*g + b over rows of 512; writes f32 stream + bf16 mirror.
__global__ __launch_bounds__(256) void ln_residual(
    const float* __restrict__ X, const float* __restrict__ Aa,
    float* __restrict__ Y, unsigned short* __restrict__ Yb,
    const float* __restrict__ g, const float* __restrict__ b)
{
  __shared__ float red[256];
  const int row = blockIdx.x;
  const int t = threadIdx.x;
  const float* xr = X  + (long)row * 512;
  const float* ar = Aa + (long)row * 512;
  const float v0 = xr[t] + ar[t];
  const float v1 = xr[t + 256] + ar[t + 256];
  red[t] = v0 + v1; __syncthreads();
  for (int off = 128; off > 0; off >>= 1) { if (t < off) red[t] += red[t + off]; __syncthreads(); }
  const float mean = red[0] * (1.0f / 512.0f); __syncthreads();
  const float d0 = v0 - mean, d1 = v1 - mean;
  red[t] = d0 * d0 + d1 * d1; __syncthreads();
  for (int off = 128; off > 0; off >>= 1) { if (t < off) red[t] += red[t + off]; __syncthreads(); }
  const float rstd = rsqrtf(red[0] * (1.0f / 512.0f) + 1e-5f);
  const float o0 = d0 * rstd * g[t]       + b[t];
  const float o1 = d1 * rstd * g[t + 256] + b[t + 256];
  Y[(long)row * 512 + t]        = o0;
  Y[(long)row * 512 + t + 256]  = o1;
  Yb[(long)row * 512 + t]       = f2bf(o0);
  Yb[(long)row * 512 + t + 256] = f2bf(o1);
}

// One-hot "matmul" as argmax+gather: out[row,d] = sqrt(512)*V[d,tok] + pe[s,d]
__global__ __launch_bounds__(256) void embed_gather(
    const float* __restrict__ onehot, const float* __restrict__ Vm,
    const float* __restrict__ pe, float* __restrict__ out,
    unsigned short* __restrict__ outb)
{
  __shared__ int tok;
  const int row = blockIdx.x;            // b*128 + s
  const int t = threadIdx.x;
  if (t == 0) tok = 0;
  __syncthreads();
  for (int i = t; i < 32000; i += 256)
    if (onehot[(long)row * 32000 + i] > 0.5f) tok = i;
  __syncthreads();
  const int s = row & 127;
  const int tk = tok;
  for (int d = t; d < 512; d += 256) {
    const float v = 22.627416997969522f * Vm[(long)d * 32000 + tk] + pe[s * 512 + d];
    out[(long)row * 512 + d]  = v;
    outb[(long)row * 512 + d] = f2bf(v);
  }
}

// ---------------------------------------------------------------------------
// Host orchestration
// ---------------------------------------------------------------------------
static void gemm(hipStream_t st, const unsigned short* A, const unsigned short* B,
                 void* C, const float* bias, int M, int N, int K,
                 int lda, int ldb, int ldc,
                 long sA1, long sB1, long sC1, int nb1,
                 long sA2, long sB2, long sC2, int nb2,
                 float alpha, int flags)
{
  dim3 grid(N / 64, M / 64, nb1 * nb2);
  gemm_wmma_bf16<<<grid, 128, 0, st>>>(A, B, C, bias, M, N, K, lda, ldb, ldc,
                                       sA1, sB1, sC1, nb1, sA2, sB2, sC2,
                                       alpha, flags);
}

static void cvt(hipStream_t st, const float* in, unsigned short* out, long n) {
  cvt_f32_bf16<<<(unsigned)(n / 1024), 256, 0, st>>>(in, out);
}

extern "C" void kernel_launch(void* const* d_in, const int* in_sizes, int n_in,
                              void* d_out, int out_size, void* d_ws, size_t ws_size,
                              hipStream_t stream) {
  (void)in_sizes; (void)n_in; (void)out_size; (void)ws_size;
  // Input order: jax tree_flatten (sorted dict keys at every level).
  const float* EN   = (const float*)d_in[0];
  const float* FR   = (const float*)d_in[1];
  const float* Vm   = (const float*)d_in[2];    // [512, 32000]
  const float* dW1  = (const float*)d_in[3];
  const float* dW2  = (const float*)d_in[4];
  const float* db1  = (const float*)d_in[5];
  const float* db2  = (const float*)d_in[6];
  const float* cWk  = (const float*)d_in[7];
  const float* cWo  = (const float*)d_in[8];
  const float* cWq  = (const float*)d_in[9];
  const float* cWv  = (const float*)d_in[10];
  const float* dl1b = (const float*)d_in[11];
  const float* dl1g = (const float*)d_in[12];
  const float* dl2b = (const float*)d_in[13];
  const float* dl2g = (const float*)d_in[14];
  const float* dl3b = (const float*)d_in[15];
  const float* dl3g = (const float*)d_in[16];
  const float* sWk  = (const float*)d_in[17];
  const float* sWo  = (const float*)d_in[18];
  const float* sWq  = (const float*)d_in[19];
  const float* sWv  = (const float*)d_in[20];
  const float* eW1  = (const float*)d_in[21];
  const float* eW2  = (const float*)d_in[22];
  const float* eWk  = (const float*)d_in[23];
  const float* eWo  = (const float*)d_in[24];
  const float* eWq  = (const float*)d_in[25];
  const float* eWv  = (const float*)d_in[26];
  const float* eb1  = (const float*)d_in[27];
  const float* eb2  = (const float*)d_in[28];
  const float* el1b = (const float*)d_in[29];
  const float* el1g = (const float*)d_in[30];
  const float* el2b = (const float*)d_in[31];
  const float* el2g = (const float*)d_in[32];
  const float* pe   = (const float*)d_in[33];   // [512, 512]

  // workspace partition -------------------------------------------------------
  float* xf   = (float*)d_ws;        // encoder stream f32 [1024,512]
  float* yf   = xf + 524288;         // decoder stream f32
  float* sc   = yf + 524288;         // scores f32 [64,128,128]
  float* proj = sc + 1048576;        // block output f32 [1024,512]
  unsigned short* xb    = (unsigned short*)(proj + 524288);
  unsigned short* yb    = xb + 524288;
  unsigned short* qb    = yb + 524288;       // [H,B,128,64] bf16
  unsigned short* kb    = qb + 524288;
  unsigned short* vb    = kb + 524288;
  unsigned short* pb    = vb + 524288;       // softmax probs bf16 [64,128,128]
  unsigned short* attnb = pb + 1048576;      // head-concat bf16 [1024,512]
  unsigned short* ffb   = attnb + 524288;    // relu(ff1) bf16 [1024,2048]
  unsigned short* wst   = ffb + 2097152;     // weight staging bf16 (<= 16.384M)

  const float isq = 0.125f;          // 1/sqrt(64)
  const long WHD = 8L * 512 * 64;    // per-layer [h,d,dk] weight stride
  const long WOO = 512L * 512;
  unsigned short* wq_s = wst;
  unsigned short* wk_s = wst + 262144;
  unsigned short* wv_s = wst + 524288;
  unsigned short* wo_s = wst + 786432;
  unsigned short* w1_s = wst;                // FFN reuses staging (stream-ordered)
  unsigned short* w2_s = wst + 1048576;

  auto attention = [&](const unsigned short* qin, const unsigned short* kin,
                       const float* Wq, const float* Wk, const float* Wv,
                       const float* Wo, int causal, float* stream_f32,
                       unsigned short* stream_b16,
                       const float* lng, const float* lnb) {
    cvt(stream, Wq, wq_s, 262144);
    cvt(stream, Wk, wk_s, 262144);
    cvt(stream, Wv, wv_s, 262144);
    cvt(stream, Wo, wo_s, 262144);
    // Q/K/V projections: inner batch over 8 heads -> [H,B,S,64] bf16
    gemm(stream, qin, wq_s, qb, nullptr, 1024, 64, 512, 512, 64, 64,
         0, 32768, 65536, 8, 0, 0, 0, 1, 1.0f, FLAG_CBF16);
    gemm(stream, kin, wk_s, kb, nullptr, 1024, 64, 512, 512, 64, 64,
         0, 32768, 65536, 8, 0, 0, 0, 1, 1.0f, FLAG_CBF16);
    gemm(stream, kin, wv_s, vb, nullptr, 1024, 64, 512, 512, 64, 64,
         0, 32768, 65536, 8, 0, 0, 0, 1, 1.0f, FLAG_CBF16);
    // scores = Q @ K^T / 8 over 64 (h,b) batches -> f32
    gemm(stream, qb, kb, sc, nullptr, 128, 128, 64, 64, 64, 128,
         8192, 8192, 16384, 64, 0, 0, 0, 1, isq, FLAG_TRANSB);
    softmax128<<<64 * 128, 128, 0, stream>>>(sc, pb, causal);
    // O = P @ V, stored head-concatenated bf16: C[b,s,h*64+dv]
    gemm(stream, pb, vb, attnb, nullptr, 128, 64, 128, 128, 64, 512,
         16384, 8192, 65536, 8, 131072, 65536, 64, 8, 1.0f, FLAG_CBF16);
    // output projection -> f32
    gemm(stream, attnb, wo_s, proj, nullptr, 1024, 512, 512, 512, 512, 512,
         0, 0, 0, 1, 0, 0, 0, 1, 1.0f, 0);
    ln_residual<<<1024, 256, 0, stream>>>(stream_f32, proj, stream_f32,
                                          stream_b16, lng, lnb);
  };

  auto ffn = [&](const unsigned short* inb, const float* W1, const float* B1,
                 const float* W2, const float* B2, float* stream_f32,
                 unsigned short* stream_b16, const float* lng, const float* lnb) {
    cvt(stream, W1, w1_s, 1048576);
    cvt(stream, W2, w2_s, 1048576);
    gemm(stream, inb, w1_s, ffb, B1, 1024, 2048, 512, 512, 2048, 2048,
         0, 0, 0, 1, 0, 0, 0, 1, 1.0f, FLAG_BIAS | FLAG_RELU | FLAG_CBF16);
    gemm(stream, ffb, w2_s, proj, B2, 1024, 512, 2048, 2048, 512, 512,
         0, 0, 0, 1, 0, 0, 0, 1, 1.0f, FLAG_BIAS);
    ln_residual<<<1024, 256, 0, stream>>>(stream_f32, proj, stream_f32,
                                          stream_b16, lng, lnb);
  };

  // ---------------- encoder ----------------
  embed_gather<<<1024, 256, 0, stream>>>(EN, Vm, pe, xf, xb);
  for (int l = 0; l < 6; ++l) {
    attention(xb, xb, eWq + l * WHD, eWk + l * WHD, eWv + l * WHD, eWo + l * WOO,
              0, xf, xb, el1g + l * 512, el1b + l * 512);
    ffn(xb, eW1 + (long)l * 1048576, eb1 + l * 2048,
        eW2 + (long)l * 1048576, eb2 + l * 512, xf, xb,
        el2g + l * 512, el2b + l * 512);
  }

  // ---------------- decoder ----------------
  embed_gather<<<1024, 256, 0, stream>>>(FR, Vm, pe, yf, yb);
  for (int l = 0; l < 6; ++l) {
    attention(yb, yb, sWq + l * WHD, sWk + l * WHD, sWv + l * WHD, sWo + l * WOO,
              1, yf, yb, dl1g + l * 512, dl1b + l * 512);
    attention(yb, xb, cWq + l * WHD, cWk + l * WHD, cWv + l * WHD, cWo + l * WOO,
              0, yf, yb, dl2g + l * 512, dl2b + l * 512);
    ffn(yb, dW1 + (long)l * 1048576, db1 + l * 2048,
        dW2 + (long)l * 1048576, db2 + l * 512, yf, yb,
        dl3g + l * 512, dl3b + l * 512);
  }

  // ---------------- logits, stored directly as [B, vocab, S] ----------------
  cvt(stream, Vm, wst, 16384000);
  gemm(stream, yb, wst, (float*)d_out, nullptr, 1024, 32000, 512, 512, 32000, 32000,
       0, 0, 0, 1, 0, 0, 0, 1, 1.0f, FLAG_PERM);
}
